// RecurrentRepresentationAggregator_28475633172680
// MI455X (gfx1250) — matrix operand order repaired and checked by reading, not compiled, not run on hardware
//
#include <hip/hip_runtime.h>
#include <hip/hip_bf16.h>
#include <stddef.h>

// ---------------------------------------------------------------------------
// Types for CDNA5 WMMA (wave32): v_wmma_f32_16x16x32_bf16
// ---------------------------------------------------------------------------
typedef __attribute__((ext_vector_type(16))) __bf16 v16bf;
typedef __attribute__((ext_vector_type(8)))  __bf16 v8bf;
typedef __attribute__((ext_vector_type(8)))  float  v8f;

// Problem constants
#define BATCH   512
#define TSTEPS  64
#define C_IN    512
#define C_HID   512
#define C_OUT   256
#define K_RH    768                 // r(256) + h(512)
#define K_ALL   1280
#define N_GATES 2048

// Workspace layout (bytes, all 256B aligned)
#define WR_OFF   0u                                   // 2048*1280 bf16
#define WF_OFF   5242880u                             // 256*512  bf16
#define Z_OFF    5505024u                             // 512*64*512 bf16
#define IN0_OFF  39059456u                            // 512*768 bf16 (ping)
#define IN1_OFF  39845888u                            // 512*768 bf16 (pong)

__device__ __forceinline__ __bf16 f2bf(float f) {
  unsigned u = __builtin_bit_cast(unsigned, f);
  unsigned r = (u + 0x7FFFu + ((u >> 16) & 1u)) >> 16;   // round-nearest-even
  unsigned short s = (unsigned short)r;
  return __builtin_bit_cast(__bf16, s);
}

__device__ __forceinline__ float sigm(float x) {
  return 1.0f / (1.0f + __expf(-x));
}

// B fragment: 16 contiguous bf16 along K (lane holds N=lq, K=kk+half*16+e)
__device__ __forceinline__ v16bf load_b16(const __bf16* p) {
  const v8bf* q = (const v8bf*)p;
  v8bf lo = q[0];
  v8bf hi = q[1];
  v16bf r;
#pragma unroll
  for (int e = 0; e < 8; ++e) { r[e] = lo[e]; r[8 + e] = hi[e]; }
  return r;
}

// A fragment: per-lane row, element e -> k = kk + half*16 + e
__device__ __forceinline__ v16bf load_a(const __bf16* row, int kk, int half) {
  const v8bf* p0 = (const v8bf*)(row + kk + half * 16);
  const v8bf* p1 = (const v8bf*)(row + kk + half * 16 + 8);
  v8bf a0 = p0[0];
  v8bf a1 = p1[0];
  v16bf r;
#pragma unroll
  for (int e = 0; e < 8; ++e) { r[e] = a0[e]; r[8 + e] = a1[e]; }
  return r;
}

// ---------------------------------------------------------------------------
// Prep: fp32 -> bf16 conversion
// ---------------------------------------------------------------------------
__global__ void k_cvt_bf16(const float* __restrict__ src,
                           __bf16* __restrict__ dst, int n) {
  int i = blockIdx.x * blockDim.x + threadIdx.x;
  if (i < n) dst[i] = f2bf(src[i]);
}

// Init state: r,h,c fp32 into d_out regions; bf16 [r,h] into ping buffer
__global__ void k_init(const float* __restrict__ r0, const float* __restrict__ h0,
                       const float* __restrict__ c0, float* __restrict__ r,
                       float* __restrict__ h, float* __restrict__ c,
                       __bf16* __restrict__ in0) {
  int i = blockIdx.x * blockDim.x + threadIdx.x;   // 0 .. 512*512-1
  if (i < BATCH * C_HID) {
    float hv = h0[i];
    h[i] = hv;
    c[i] = c0[i];
    int b = i / C_HID, j = i % C_HID;
    in0[(size_t)b * K_RH + C_OUT + j] = f2bf(hv);
  }
  if (i < BATCH * C_OUT) {
    float rv = r0[i];
    r[i] = rv;
    int b = i / C_OUT, n = i % C_OUT;
    in0[(size_t)b * K_RH + n] = f2bf(rv);
  }
}

// ---------------------------------------------------------------------------
// Kernel A: gates GEMM + fused LSTM pointwise.
//   Each wave computes a 32(M) x 16(J) output patch for ALL 4 gates:
//   2 A fragments (adjacent 16-row M-tiles), each B fragment feeds 2 WMMAs.
//   512 waves = 16 M-pairs x 32 J-tiles -> 64 blocks x 256 threads (8 waves).
// ---------------------------------------------------------------------------
__global__ void __launch_bounds__(256)
k_gates_lstm(const __bf16* __restrict__ in_cur,   // [512, 768] bf16 [r|h]
             __bf16* __restrict__ in_next,        // [512, 768] bf16 (h part written)
             const __bf16* __restrict__ zb,       // [512, 64, 512] bf16
             const __bf16* __restrict__ Wr,       // [2048, 1280] bf16
             const float* __restrict__ b_rnn,     // [2048]
             float* __restrict__ h,               // [512, 512]
             float* __restrict__ c,               // [512, 512]
             int t) {
  const int wave  = blockIdx.x * 8 + (threadIdx.x >> 5);
  const int lane  = threadIdx.x & 31;
  const int half  = lane >> 4;
  const int lq    = lane & 15;
  const int mpair = wave >> 5;           // 0..15 (two 16-row M-tiles)
  const int jtile = wave & 31;           // 0..31

  const int m0 = mpair * 32 + lq;        // A row, first M-tile
  const int m1 = m0 + 16;                // A row, second M-tile
  const int j  = jtile * 16 + lq;        // output hidden column for this lane

  const __bf16* arow_rh0 = in_cur + (size_t)m0 * K_RH;
  const __bf16* arow_rh1 = in_cur + (size_t)m1 * K_RH;
  const __bf16* arow_z0  = zb + ((size_t)m0 * TSTEPS + t) * C_IN;
  const __bf16* arow_z1  = zb + ((size_t)m1 * TSTEPS + t) * C_IN;

  const __bf16* wrow[4];
#pragma unroll
  for (int g = 0; g < 4; ++g)
    wrow[g] = Wr + (size_t)(g * C_HID + j) * K_ALL + half * 16;

  v8f acc[4][2];
#pragma unroll
  for (int g = 0; g < 4; ++g)
#pragma unroll
    for (int p = 0; p < 2; ++p)
      acc[g][p] = (v8f){0.f, 0.f, 0.f, 0.f, 0.f, 0.f, 0.f, 0.f};

  // K over [r|h]: 24 blocks of 32
  for (int kk = 0; kk < K_RH; kk += 32) {
    v16bf a0 = load_a(arow_rh0, kk, half);
    v16bf a1 = load_a(arow_rh1, kk, half);
#pragma unroll
    for (int g = 0; g < 4; ++g) {
      v16bf b = load_b16(wrow[g] + kk);
      acc[g][0] = __builtin_amdgcn_wmma_f32_16x16x32_bf16(
          false, a0, false, b, (short)0, acc[g][0], false, false);
      acc[g][1] = __builtin_amdgcn_wmma_f32_16x16x32_bf16(
          false, a1, false, b, (short)0, acc[g][1], false, false);
    }
  }
  // K over z_t: 16 blocks of 32
  for (int kz = 0; kz < C_IN; kz += 32) {
    v16bf a0 = load_a(arow_z0, kz, half);
    v16bf a1 = load_a(arow_z1, kz, half);
#pragma unroll
    for (int g = 0; g < 4; ++g) {
      v16bf b = load_b16(wrow[g] + K_RH + kz);
      acc[g][0] = __builtin_amdgcn_wmma_f32_16x16x32_bf16(
          false, a0, false, b, (short)0, acc[g][0], false, false);
      acc[g][1] = __builtin_amdgcn_wmma_f32_16x16x32_bf16(
          false, a1, false, b, (short)0, acc[g][1], false, false);
    }
  }

  // Fused LSTM pointwise. C/D layout: VGPR v, lane -> M = base + half*8 + v, N = j
  const float bf_ = b_rnn[j];
  const float bi_ = b_rnn[C_HID + j];
  const float bs_ = b_rnn[2 * C_HID + j];
  const float bo_ = b_rnn[3 * C_HID + j];
#pragma unroll
  for (int p = 0; p < 2; ++p) {
#pragma unroll
    for (int v = 0; v < 8; ++v) {
      int mr = mpair * 32 + p * 16 + half * 8 + v;
      size_t idx = (size_t)mr * C_HID + j;
      float f = acc[0][p][v] + bf_;
      float i = acc[1][p][v] + bi_;
      float s = acc[2][p][v] + bs_;
      float o = acc[3][p][v] + bo_;
      float cn = sigm(f) * c[idx] + sigm(i) * tanhf(s);
      float hn = sigm(o) * tanhf(cn);
      c[idx] = cn;
      h[idx] = hn;
      in_next[(size_t)mr * K_RH + C_OUT + j] = f2bf(hn);
    }
  }
}

// ---------------------------------------------------------------------------
// Kernel B: r_n = r + h_n @ W_feat^T + b_feat  (M=512, N=256, K=512)
//   Each wave: 32(M) x 16(N) patch. 256 waves = 16 M-pairs x 16 N-tiles
//   -> 32 blocks x 256 threads.
// ---------------------------------------------------------------------------
__global__ void __launch_bounds__(256)
k_feat(const __bf16* __restrict__ in_next,   // [512,768]: h bf16 at +256, r bf16 written at +0
       __bf16* __restrict__ in_next_w,
       const __bf16* __restrict__ Wf,        // [256, 512] bf16
       const float* __restrict__ b_feat,     // [256]
       float* __restrict__ r) {              // [512, 256]
  const int wave  = blockIdx.x * 8 + (threadIdx.x >> 5);
  const int lane  = threadIdx.x & 31;
  const int half  = lane >> 4;
  const int lq    = lane & 15;
  const int mpair = wave >> 4;           // 0..15
  const int ntile = wave & 15;           // 0..15

  const int m0 = mpair * 32 + lq;
  const int m1 = m0 + 16;
  const int n  = ntile * 16 + lq;

  const __bf16* arow0 = in_next + (size_t)m0 * K_RH + C_OUT;      // h part, K=512
  const __bf16* arow1 = in_next + (size_t)m1 * K_RH + C_OUT;
  const __bf16* wrow  = Wf + (size_t)n * C_HID + half * 16;

  v8f acc0 = (v8f){0.f, 0.f, 0.f, 0.f, 0.f, 0.f, 0.f, 0.f};
  v8f acc1 = acc0;
  for (int kk = 0; kk < C_HID; kk += 32) {
    v16bf a0 = load_a(arow0, kk, half);
    v16bf a1 = load_a(arow1, kk, half);
    v16bf b  = load_b16(wrow + kk);
    acc0 = __builtin_amdgcn_wmma_f32_16x16x32_bf16(
        false, a0, false, b, (short)0, acc0, false, false);
    acc1 = __builtin_amdgcn_wmma_f32_16x16x32_bf16(
        false, a1, false, b, (short)0, acc1, false, false);
  }

  const float bn = b_feat[n];
#pragma unroll
  for (int p = 0; p < 2; ++p) {
    const v8f& acc = p ? acc1 : acc0;
#pragma unroll
    for (int v = 0; v < 8; ++v) {
      int mr = mpair * 32 + p * 16 + half * 8 + v;
      size_t idx = (size_t)mr * C_OUT + n;
      float rn = r[idx] + acc[v] + bn;
      r[idx] = rn;
      in_next_w[(size_t)mr * K_RH + n] = f2bf(rn);
    }
  }
}

// ---------------------------------------------------------------------------
// Host launcher
// ---------------------------------------------------------------------------
extern "C" void kernel_launch(void* const* d_in, const int* in_sizes, int n_in,
                              void* d_out, int out_size, void* d_ws, size_t ws_size,
                              hipStream_t stream) {
  const float* z      = (const float*)d_in[0];   // [512,64,512]
  const float* r0     = (const float*)d_in[1];   // [512,256]
  const float* h0     = (const float*)d_in[2];   // [512,512]
  const float* c0     = (const float*)d_in[3];   // [512,512]
  const float* W_rnn  = (const float*)d_in[4];   // [2048,1280]
  const float* b_rnn  = (const float*)d_in[5];   // [2048]
  const float* W_feat = (const float*)d_in[6];   // [256,512]
  const float* b_feat = (const float*)d_in[7];   // [256]

  // Outputs live directly in d_out: r | h | c
  float* r = (float*)d_out;
  float* h = r + BATCH * C_OUT;
  float* c = h + BATCH * C_HID;

  char* ws = (char*)d_ws;
  __bf16* Wr_bf = (__bf16*)(ws + WR_OFF);
  __bf16* Wf_bf = (__bf16*)(ws + WF_OFF);
  __bf16* z_bf  = (__bf16*)(ws + Z_OFF);
  __bf16* in_buf[2] = { (__bf16*)(ws + IN0_OFF), (__bf16*)(ws + IN1_OFF) };

  // Prep: convert weights + z to bf16, init state
  {
    int n = N_GATES * K_ALL;
    k_cvt_bf16<<<(n + 255) / 256, 256, 0, stream>>>(W_rnn, Wr_bf, n);
  }
  {
    int n = C_OUT * C_HID;
    k_cvt_bf16<<<(n + 255) / 256, 256, 0, stream>>>(W_feat, Wf_bf, n);
  }
  {
    int n = BATCH * TSTEPS * C_IN;
    k_cvt_bf16<<<(n + 255) / 256, 256, 0, stream>>>(z, z_bf, n);
  }
  {
    int n = BATCH * C_HID;
    k_init<<<(n + 255) / 256, 256, 0, stream>>>(r0, h0, c0, r, h, c, in_buf[0]);
  }

  // Recurrence: 64 serial steps, ping-pong bf16 [r|h] buffer
  for (int t = 0; t < TSTEPS; ++t) {
    __bf16* cur = in_buf[t & 1];
    __bf16* nxt = in_buf[(t + 1) & 1];
    k_gates_lstm<<<64, 256, 0, stream>>>(cur, nxt, z_bf, Wr_bf, b_rnn, h, c, t);
    k_feat<<<32, 256, 0, stream>>>(nxt, nxt, Wf_bf, b_feat, r);
  }
}
